// MultiHeadAttention_26233660244239
// MI455X (gfx1250) — matrix operand order repaired and checked by reading, not compiled, not run on hardware
//
#include <hip/hip_runtime.h>
#include <hip/hip_bf16.h>
#include <math.h>

// ---------------------------------------------------------------------------
// MHA forward for gfx1250 (MI455X): bf16 WMMA GEMMs + flash attention,
// with Tensor Data Mover (TDM) double-buffered LDS staging.
// B=4, S=2048, E=1024, H=16, D=64.
// ---------------------------------------------------------------------------

typedef __bf16 bf16_t;
typedef __attribute__((ext_vector_type(8)))  __bf16 v8bf;
typedef __attribute__((ext_vector_type(16))) __bf16 v16bf;
typedef __attribute__((ext_vector_type(8)))  float  v8f;
typedef __attribute__((ext_vector_type(4)))  unsigned int u32x4;
typedef __attribute__((ext_vector_type(4)))  int i32x4;
typedef __attribute__((ext_vector_type(8)))  int i32x8;

#define BDIM 4
#define SDIM 2048
#define EDIM 1024
#define HDIM 16
#define DDIM 64
#define MDIM (BDIM * SDIM)   // 8192 rows in the flattened GEMMs
#define LROW 72              // LDS row stride (64 + 8 pad, keeps 16B align)

#define ATILE (128 * LROW)   // A-tile elems in LDS
#define BTILE (64 * LROW)    // B/K/V-tile elems in LDS

#if __has_builtin(__builtin_amdgcn_tensor_load_to_lds)
#define HAS_TDM 1
#else
#define HAS_TDM 0
#endif

static __device__ __forceinline__ v16bf cat16(v8bf lo, v8bf hi) {
  return __builtin_shufflevector(lo, hi, 0,1,2,3,4,5,6,7,8,9,10,11,12,13,14,15);
}

static __device__ __forceinline__ v8f wmma_bf16(v16bf a, v16bf b, v8f c) {
  return __builtin_amdgcn_wmma_f32_16x16x32_bf16(
      /*neg_a=*/false, a, /*neg_b=*/false, b,
      /*c_mod=*/(short)0, c, /*reuse_a=*/false, /*reuse_b=*/false);
}

// LDS byte offset of a generic pointer into __shared__ space: the generic
// address is {SHARED_BASE[63:32], lds_offset[31:0]}, so truncation works.
static __device__ __forceinline__ unsigned lds_off_of(const void* p) {
  return (unsigned)(unsigned long long)p;
}

// ---------------------------------------------------------------------------
// TDM: load a 2-D bf16 tile (64 elems wide x `rows` high, global row stride
// `stride_elems`) into LDS with 16B padding appended to every 128B row, so
// the LDS image has LROW=72 element rows. Issued per-wave (EXEC ignored);
// callers gate on wave id and track completion with TENSORcnt.
// D# layout per CDNA5 ISA 8.3/8.4.
// ---------------------------------------------------------------------------
static __device__ __forceinline__ void tdm_load_tile64(
    unsigned lds_byte, const void* gptr, int rows, int stride_elems) {
#if HAS_TDM
  unsigned long long ga = (unsigned long long)gptr;
  u32x4 g0;
  g0[0] = 1u;                                   // count=1 (valid descriptor)
  g0[1] = lds_byte;                             // lds_addr
  g0[2] = (unsigned)ga;                         // global_addr[31:0]
  g0[3] = (unsigned)((ga >> 32) & 0x1FFFFFFull) // global_addr[56:32]
          | (2u << 30);                         // type = 2 ("image")
  i32x8 g1;
  g1[0] = (int)((1u << 16)    // data_size = 2 bytes
              | (1u << 20)    // pad_enable
              | (4u << 22)    // pad_interval: every 32 dwords (=64 bf16)
              | (3u << 25));  // pad_amount: 4 dwords (=16B)
  g1[1] = (int)(64u << 16);             // tensor_dim0 = 64
  g1[2] = (int)((unsigned)rows << 16);  // tensor_dim1[15:0] = rows
  g1[3] = (int)(64u << 16);             // tile_dim0 = 64
  g1[4] = rows;                         // tile_dim1 = rows
  g1[5] = stride_elems;                 // tensor_dim0_stride[31:0]
  g1[6] = 0;
  g1[7] = 0;
  i32x4 gz = {0, 0, 0, 0};
#if defined(__clang_major__) && (__clang_major__ >= 23)
  i32x8 gz8 = {0, 0, 0, 0, 0, 0, 0, 0};
  __builtin_amdgcn_tensor_load_to_lds(g0, g1, gz, gz, gz8, 0);
#else
  __builtin_amdgcn_tensor_load_to_lds(g0, g1, gz, gz, 0);
#endif
#else
  (void)lds_byte; (void)gptr; (void)rows; (void)stride_elems;
#endif
}

static __device__ __forceinline__ void tdm_wait0() {
#if HAS_TDM
  __builtin_amdgcn_s_wait_tensorcnt(0);
#endif
}

// ---------------------------------------------------------------------------
// fp32 -> bf16 convert (grid-stride)
// ---------------------------------------------------------------------------
__global__ void __launch_bounds__(256)
cvt_f32_bf16(const float* __restrict__ in, bf16_t* __restrict__ out, int n) {
  int i = blockIdx.x * blockDim.x + threadIdx.x;
  int stride = gridDim.x * blockDim.x;
  for (; i < n; i += stride) out[i] = (bf16_t)in[i];
}

// ---------------------------------------------------------------------------
// Tiled GEMM: Y[m,n] = sum_k A[m,k] * W[n,k]   (torch Linear, W is [out,in])
// Tile: 128(M) x 64(N) per workgroup, 8 waves, each wave owns 16 rows.
// TDM double-buffered K staging: issue tile k+1, compute tile k, wait, barrier.
// MODE 0: bf16 out, layout [b,h,s,d], scaled      (Q and K projections)
// MODE 2: bf16 out, layout [b,h,d,s] (transposed) (V projection)
// MODE 3: fp32 out + bias, layout [m,n]           (output projection)
// ---------------------------------------------------------------------------
template <int MODE>
__global__ void __launch_bounds__(256)
gemm_wmma(const bf16_t* __restrict__ A, const bf16_t* __restrict__ W,
          bf16_t* __restrict__ outb, float* __restrict__ outf,
          const float* __restrict__ bias, float scale) {
  __shared__ bf16_t smem[2 * ATILE + 2 * BTILE];  // 54KB
  bf16_t* const asbuf = smem;              // 2 x 128x72 A tiles
  bf16_t* const bsbuf = smem + 2 * ATILE;  // 2 x  64x72 B tiles

  const int mBase = blockIdx.y * 128;
  const int nBase = blockIdx.x * 64;
  const int tid  = threadIdx.x;
  const int lane = tid & 31;
  const int wv   = tid >> 5;       // wave 0..7
  const int kg   = lane >> 4;      // lane half
  const int col  = lane & 15;

  v8f acc[4];
  const v8f z8 = {0.f, 0.f, 0.f, 0.f, 0.f, 0.f, 0.f, 0.f};
#pragma unroll
  for (int t = 0; t < 4; ++t) acc[t] = z8;

#if HAS_TDM
  if (wv == 0) {  // uniform per-wave branch; only wave0 issues DMA
    tdm_load_tile64(lds_off_of(asbuf), &A[(size_t)mBase * EDIM], 128, EDIM);
    tdm_load_tile64(lds_off_of(bsbuf), &W[(size_t)nBase * EDIM], 64, EDIM);
    tdm_wait0();
  }
  __syncthreads();
#endif

  for (int k0 = 0; k0 < EDIM; k0 += 64) {
    const int cur = (k0 >> 6) & 1;
    bf16_t* const as = asbuf + cur * ATILE;
    bf16_t* const bs = bsbuf + cur * BTILE;
#if HAS_TDM
    if (k0 + 64 < EDIM && wv == 0) {  // kick off next tile while computing
      tdm_load_tile64(lds_off_of(asbuf + (cur ^ 1) * ATILE),
                      &A[(size_t)mBase * EDIM + k0 + 64], 128, EDIM);
      tdm_load_tile64(lds_off_of(bsbuf + (cur ^ 1) * BTILE),
                      &W[(size_t)nBase * EDIM + k0 + 64], 64, EDIM);
    }
#else
    // Fallback: manual staging into the current buffer.
#pragma unroll
    for (int i = 0; i < 4; ++i) {
      int v = tid + i * 256;
      int row = v >> 3, c8 = (v & 7) * 8;
      *(uint4*)&as[row * LROW + c8] =
          *(const uint4*)&A[(size_t)(mBase + row) * EDIM + k0 + c8];
    }
#pragma unroll
    for (int i = 0; i < 2; ++i) {
      int v = tid + i * 256;
      int row = v >> 3, c8 = (v & 7) * 8;
      *(uint4*)&bs[row * LROW + c8] =
          *(const uint4*)&W[(size_t)(nBase + row) * EDIM + k0 + c8];
    }
    __syncthreads();
#endif

    const int ra = wv * 16 + (lane & 15);  // A-fragment row (per lane)
#pragma unroll
    for (int ks = 0; ks < 2; ++ks) {
      v8bf alo = *(const v8bf*)&as[ra * LROW + ks * 32 + kg * 8];
      v8bf ahi = *(const v8bf*)&as[ra * LROW + ks * 32 + 16 + kg * 8];
      v16bf af = cat16(alo, ahi);
#pragma unroll
      for (int t = 0; t < 4; ++t) {
        v8bf blo = *(const v8bf*)&bs[(t * 16 + col) * LROW + ks * 32 + kg * 8];
        v8bf bhi = *(const v8bf*)&bs[(t * 16 + col) * LROW + ks * 32 + 16 + kg * 8];
        acc[t] = wmma_bf16(af, cat16(blo, bhi), acc[t]);
      }
    }
#if HAS_TDM
    if (wv == 0) tdm_wait0();  // next buffer landed
#endif
    __syncthreads();
  }

  // Epilogue: C/D layout -> VGPR r holds M=r (lanes 0-15) / M=r+8 (16-31).
  const int rbase = kg * 8;
#pragma unroll
  for (int t = 0; t < 4; ++t) {
#pragma unroll
    for (int r = 0; r < 8; ++r) {
      int m = mBase + wv * 16 + rbase + r;
      int n = nBase + t * 16 + col;
      float val = acc[t][r] * scale;
      if (MODE == 0) {
        int b = m >> 11, s = m & (SDIM - 1), h = n >> 6, d = n & 63;
        outb[(((size_t)(b * HDIM + h) * SDIM + s) * DDIM + d)] = (bf16_t)val;
      } else if (MODE == 2) {
        int b = m >> 11, s = m & (SDIM - 1), h = n >> 6, d = n & 63;
        outb[(((size_t)(b * HDIM + h) * DDIM + d) * SDIM + s)] = (bf16_t)val;
      } else {
        outf[(size_t)m * EDIM + n] = val + bias[n];
      }
    }
  }
}

// ---------------------------------------------------------------------------
// Flash attention: one (b,h) x 128-query tile per workgroup. 8 waves, each
// wave owns 16 query rows. 64-wide key tiles, causal-truncated, TDM
// double-buffered. The Q staging buffer is dead once Q fragments are in
// registers; each wave's P re-layout buffer aliases exactly the 16 Q rows
// that same wave consumed (same-wave LDS ops are in-order -> no barrier).
// Q: [b,h,s,d] bf16 (pre-scaled 1/sqrt(D)), K: [b,h,s,d], V: [b,h,d,s].
// ---------------------------------------------------------------------------
__global__ void __launch_bounds__(256)
flash_attn(const bf16_t* __restrict__ Q, const bf16_t* __restrict__ K,
           const bf16_t* __restrict__ V, bf16_t* __restrict__ O) {
  __shared__ bf16_t smem[ATILE + 4 * BTILE];  // 54KB
  bf16_t* const Qs   = smem;                  // aliased as P after qf load
  bf16_t* const Kbuf = smem + ATILE;          // 2 x 64x72 K tiles
  bf16_t* const Vbuf = smem + ATILE + 2 * BTILE;  // 2 x 64x72 V tiles

  const int bh    = blockIdx.y;
  const int qBase = blockIdx.x * 128;
  const int tid  = threadIdx.x;
  const int lane = tid & 31;
  const int wv   = tid >> 5;
  const int kg   = lane >> 4;
  const int col  = lane & 15;
  const int rbase = kg * 8;
  bf16_t* const Pw = Qs + wv * 16 * LROW;  // this wave's 16x72 P slice

  const bf16_t* Qp = Q + (size_t)bh * SDIM * DDIM;
  const bf16_t* Kp = K + (size_t)bh * SDIM * DDIM;
  const bf16_t* Vp = V + (size_t)bh * DDIM * SDIM;
  const int nTiles = blockIdx.x * 2 + 2;  // causal: keys up to qBase+128

#if HAS_TDM
  if (wv == 0) {
    tdm_load_tile64(lds_off_of(Qs), Qp + (size_t)qBase * DDIM, 128, DDIM);
    tdm_load_tile64(lds_off_of(Kbuf), Kp, 64, DDIM);
    tdm_load_tile64(lds_off_of(Vbuf), Vp, 64, SDIM);
    tdm_wait0();
  }
  __syncthreads();
#else
#pragma unroll
  for (int i = 0; i < 4; ++i) {
    int v = tid + i * 256;
    int row = v >> 3, c8 = (v & 7) * 8;
    *(uint4*)&Qs[row * LROW + c8] =
        *(const uint4*)&Qp[(size_t)(qBase + row) * DDIM + c8];
  }
  __syncthreads();
#endif

  // Q fragments (each wave reads only its own 16 rows).
  v16bf qf[2];
  {
    const int ra = wv * 16 + (lane & 15);
#pragma unroll
    for (int ks = 0; ks < 2; ++ks) {
      v8bf lo = *(const v8bf*)&Qs[ra * LROW + ks * 32 + kg * 8];
      v8bf hi = *(const v8bf*)&Qs[ra * LROW + ks * 32 + 16 + kg * 8];
      qf[ks] = cat16(lo, hi);
    }
  }

  const v8f z8 = {0.f, 0.f, 0.f, 0.f, 0.f, 0.f, 0.f, 0.f};
  v8f oacc[4];
#pragma unroll
  for (int t = 0; t < 4; ++t) oacc[t] = z8;
  float Mrow[8], Lrow[8];
#pragma unroll
  for (int r = 0; r < 8; ++r) { Mrow[r] = -1e30f; Lrow[r] = 0.f; }

  for (int jt = 0; jt < nTiles; ++jt) {
    const int j0 = jt * 64;
    const int cur = jt & 1;
    const bf16_t* const ks_ = Kbuf + cur * BTILE;
    const bf16_t* const vs_ = Vbuf + cur * BTILE;
#if HAS_TDM
    if (jt + 1 < nTiles && wv == 0) {  // prefetch next K/V tile via TDM
      const int j1 = j0 + 64;
      tdm_load_tile64(lds_off_of(Kbuf + (cur ^ 1) * BTILE),
                      Kp + (size_t)j1 * DDIM, 64, DDIM);
      tdm_load_tile64(lds_off_of(Vbuf + (cur ^ 1) * BTILE),
                      Vp + j1, 64, SDIM);
    }
#else
#pragma unroll
    for (int i = 0; i < 2; ++i) {
      int v = tid + i * 256;
      int row = v >> 3, c8 = (v & 7) * 8;
      *(uint4*)&((bf16_t*)ks_)[row * LROW + c8] =
          *(const uint4*)&Kp[(size_t)(j0 + row) * DDIM + c8];
      *(uint4*)&((bf16_t*)vs_)[row * LROW + c8] =
          *(const uint4*)&Vp[(size_t)row * SDIM + j0 + c8];
    }
    __syncthreads();
#endif

    // S = Q K^T (Q pre-scaled). 4 N-tiles of 16, K-dim = D = 2x32.
    v8f sacc[4];
#pragma unroll
    for (int t = 0; t < 4; ++t) sacc[t] = z8;
#pragma unroll
    for (int ks = 0; ks < 2; ++ks) {
#pragma unroll
      for (int t = 0; t < 4; ++t) {
        v8bf blo = *(const v8bf*)&ks_[(t * 16 + col) * LROW + ks * 32 + kg * 8];
        v8bf bhi = *(const v8bf*)&ks_[(t * 16 + col) * LROW + ks * 32 + 16 + kg * 8];
        sacc[t] = wmma_bf16(qf[ks], cat16(blo, bhi), sacc[t]);
      }
    }

    // Causal mask + running row max (reduce over 16-lane half).
    float rmax[8];
#pragma unroll
    for (int r = 0; r < 8; ++r) {
      const int qg = qBase + wv * 16 + rbase + r;
      float mx = -1e30f;
#pragma unroll
      for (int t = 0; t < 4; ++t) {
        const int kgi = j0 + t * 16 + col;
        float s = (kgi > qg) ? -1e30f : sacc[t][r];
        sacc[t][r] = s;
        mx = fmaxf(mx, s);
      }
      rmax[r] = mx;
    }
#pragma unroll
    for (int off = 8; off; off >>= 1)
#pragma unroll
      for (int r = 0; r < 8; ++r)
        rmax[r] = fmaxf(rmax[r], __shfl_xor(rmax[r], off, 32));

    float mnew[8], corr[8], rsum[8];
#pragma unroll
    for (int r = 0; r < 8; ++r) {
      mnew[r] = fmaxf(Mrow[r], rmax[r]);
      corr[r] = __expf(Mrow[r] - mnew[r]);
      Mrow[r] = mnew[r];
    }

    // P = exp(S - m): write to this wave's LDS slice (C layout -> A layout).
#pragma unroll
    for (int r = 0; r < 8; ++r) {
      float rs = 0.f;
#pragma unroll
      for (int t = 0; t < 4; ++t) {
        float p = __expf(sacc[t][r] - mnew[r]);
        rs += p;
        Pw[(rbase + r) * LROW + t * 16 + col] = (bf16_t)p;
      }
      rsum[r] = rs;
    }
#pragma unroll
    for (int off = 8; off; off >>= 1)
#pragma unroll
      for (int r = 0; r < 8; ++r)
        rsum[r] += __shfl_xor(rsum[r], off, 32);
#pragma unroll
    for (int r = 0; r < 8; ++r) Lrow[r] = Lrow[r] * corr[r] + rsum[r];

    // Rescale running O, then O += P @ V.
#pragma unroll
    for (int t = 0; t < 4; ++t)
#pragma unroll
      for (int r = 0; r < 8; ++r) oacc[t][r] *= corr[r];

#pragma unroll
    for (int ks = 0; ks < 2; ++ks) {
      v8bf plo = *(const v8bf*)&Pw[(lane & 15) * LROW + ks * 32 + kg * 8];
      v8bf phi = *(const v8bf*)&Pw[(lane & 15) * LROW + ks * 32 + 16 + kg * 8];
      v16bf pf = cat16(plo, phi);
#pragma unroll
      for (int t = 0; t < 4; ++t) {
        v8bf vlo = *(const v8bf*)&vs_[(t * 16 + col) * LROW + ks * 32 + kg * 8];
        v8bf vhi = *(const v8bf*)&vs_[(t * 16 + col) * LROW + ks * 32 + 16 + kg * 8];
        oacc[t] = wmma_bf16(pf, cat16(vlo, vhi), oacc[t]);
      }
    }
#if HAS_TDM
    if (wv == 0) tdm_wait0();  // next K/V buffer landed
#endif
    __syncthreads();
  }

  // Normalize and store to [b, s, h*64+d] bf16.
  const int b = bh >> 4, h = bh & 15;
#pragma unroll
  for (int t = 0; t < 4; ++t) {
#pragma unroll
    for (int r = 0; r < 8; ++r) {
      const int s = qBase + wv * 16 + rbase + r;
      const int e = h * DDIM + t * 16 + col;
      O[((size_t)(b * SDIM + s)) * EDIM + e] = (bf16_t)(oacc[t][r] / Lrow[r]);
    }
  }
}

// ---------------------------------------------------------------------------
// Host-side orchestration (graph-capture safe: only kernel launches).
// ---------------------------------------------------------------------------
extern "C" void kernel_launch(void* const* d_in, const int* in_sizes, int n_in,
                              void* d_out, int out_size, void* d_ws,
                              size_t ws_size, hipStream_t stream) {
  const float* x   = (const float*)d_in[0];
  // d_in[1] is the causal mask: implied by kernel, unused.
  const float* wq  = (const float*)d_in[2];
  const float* wk  = (const float*)d_in[3];
  const float* wvv = (const float*)d_in[4];
  const float* wo  = (const float*)d_in[5];
  const float* bo  = (const float*)d_in[6];
  float* out = (float*)d_out;

  const size_t XE = (size_t)MDIM * EDIM;  // 8M elems
  const size_t WE = (size_t)EDIM * EDIM;  // 1M elems
  bf16_t* Xb  = (bf16_t*)d_ws;
  bf16_t* Wqb = Xb + XE;
  bf16_t* Wkb = Wqb + WE;
  bf16_t* Wvb = Wkb + WE;
  bf16_t* Wob = Wvb + WE;
  bf16_t* Qb  = Wob + WE;
  bf16_t* Kb  = Qb + XE;
  bf16_t* Vb  = Kb + XE;
  bf16_t* Ob  = Vb + XE;

  cvt_f32_bf16<<<2048, 256, 0, stream>>>(x,   Xb,  (int)XE);
  cvt_f32_bf16<<<512,  256, 0, stream>>>(wq,  Wqb, (int)WE);
  cvt_f32_bf16<<<512,  256, 0, stream>>>(wk,  Wkb, (int)WE);
  cvt_f32_bf16<<<512,  256, 0, stream>>>(wvv, Wvb, (int)WE);
  cvt_f32_bf16<<<512,  256, 0, stream>>>(wo,  Wob, (int)WE);

  dim3 gg(EDIM / 64, MDIM / 128);  // (16, 64)
  dim3 blk(256);
  const float qscale = 0.125f;  // 1/sqrt(D), folded into Q
  gemm_wmma<0><<<gg, blk, 0, stream>>>(Xb, Wqb, Qb, nullptr, nullptr, qscale);
  gemm_wmma<0><<<gg, blk, 0, stream>>>(Xb, Wkb, Kb, nullptr, nullptr, 1.0f);
  gemm_wmma<2><<<gg, blk, 0, stream>>>(Xb, Wvb, Vb, nullptr, nullptr, 1.0f);

  flash_attn<<<dim3(SDIM / 128, BDIM * HDIM), blk, 0, stream>>>(Qb, Kb, Vb, Ob);

  gemm_wmma<3><<<gg, blk, 0, stream>>>(Ob, Wob, nullptr, out, bo, 1.0f);
}
